// LINDAAttenAgent_31722628448662
// MI455X (gfx1250) — matrix operand order repaired and checked by reading, not compiled, not run on hardware
//
#include <hip/hip_runtime.h>
#include <hip/hip_bf16.h>

typedef __attribute__((ext_vector_type(16))) _Float16 v16h;
typedef __attribute__((ext_vector_type(8)))  float    v8f;

#define DEVI __device__ __forceinline__

// ---------------- CDNA5 async global->LDS staging (ASYNCcnt-tracked) ----------------
// LDS address = low 32 bits of the generic pointer (ISA 10.2: flat->LDS truncates to addr[31:0]).
DEVI void async_ld_b128(const _Float16* __restrict__ g, _Float16* l) {
  unsigned lds = (unsigned)(unsigned long long)(uintptr_t)l;
  unsigned long long ga = (unsigned long long)(uintptr_t)g;
  asm volatile("global_load_async_to_lds_b128 %0, %1, off"
               :: "v"(lds), "v"(ga)
               : "memory");
}
DEVI void wait_async() {
  asm volatile("s_wait_asynccnt 0" ::: "memory");
}
// stage n128 16-byte chunks (per 256-thread workgroup)
DEVI void stage_w_async(const _Float16* __restrict__ g, _Float16* l, int n128) {
  for (int c = threadIdx.x; c < n128; c += 256)
    async_ld_b128(g + c * 8, l + c * 8);
}

// ---------------- WMMA fragment helpers (CDNA5 16x16x32 f16, wave32) ----------------

// A fragment from an LDS tile, row-major f16, base points at A[0][k0], row stride lda (halves).
// Layout: lane = (half, m); holds K = half*16 + 0..15 (contiguous, matching frag_b_t).
DEVI v16h frag_a_lds(const _Float16* base, int lda) {
  const int lane = threadIdx.x & 31;
  const int half = lane >> 4;
  const int m    = lane & 15;
  const _Float16* p0 = base + m * lda + half * 16;
  const _Float16* p1 = p0 + 8;
  v16h f;
#pragma unroll
  for (int i = 0; i < 8; ++i) { f[i] = p0[i]; f[8 + i] = p1[i]; }
  return f;
}

// B fragment from a transposed weight matrix Wt[N][K] (row-major, f16; global or LDS).
// lane = (half, n); holds K = k0 + half*16 + 0..15 for column n0+n -> 16 contiguous halves.
DEVI v16h frag_b_t(const _Float16* __restrict__ Wt, int ldk, int n0, int k0) {
  const int lane = threadIdx.x & 31;
  const int half = lane >> 4;
  const int n    = lane & 15;
  const _Float16* p = Wt + (size_t)(n0 + n) * ldk + k0 + half * 16;
  v16h f;
#pragma unroll
  for (int i = 0; i < 16; ++i) f[i] = p[i];
  return f;
}

DEVI v8f wmma16(v16h a, v16h b, v8f c) {
  return __builtin_amdgcn_wmma_f32_16x16x32_f16(false, a, false, b, (short)0, c, false, false);
}

// ---------------- weight prep: W[K][N] f32 -> Wt[Npad][K] f16 (zero padded) ----------------
__global__ void k_wt(const float* __restrict__ W, _Float16* __restrict__ Wt,
                     int K, int N, int Npad) {
  int idx = blockIdx.x * 256 + threadIdx.x;
  int total = Npad * K;
  if (idx >= total) return;
  int n = idx / K, k = idx % K;
  float v = (n < N) ? W[(size_t)k * N + n] : 0.0f;
  Wt[(size_t)n * K + k] = (_Float16)v;
}

// ---------------- K1: mlp1 + GRU cell ----------------
__global__ __launch_bounds__(256)
void k_gru(const float* __restrict__ inp, const float* __restrict__ h0,
           const _Float16* __restrict__ wt1, const float* __restrict__ b1,
           const _Float16* __restrict__ wtih, const float* __restrict__ bih,
           const _Float16* __restrict__ wthh, const float* __restrict__ bhh,
           float* __restrict__ hout) {
  __shared__ alignas(16) _Float16 sIn[16][96];
  __shared__ alignas(16) _Float16 sX[16][64];
  __shared__ alignas(16) _Float16 sH[16][64];
  __shared__ alignas(16) float    sG[16][384];   // gx cols 0..191, gh cols 192..383
  const int r0 = blockIdx.x * 16;
  const int tid = threadIdx.x;
  for (int e = tid; e < 16 * 96; e += 256) {
    int l = e / 96, c = e % 96;
    sIn[l][c] = (_Float16)inp[(size_t)(r0 + l) * 96 + c];
  }
  for (int e = tid; e < 16 * 64; e += 256) {
    int l = e >> 6, c = e & 63;
    sH[l][c] = (_Float16)h0[(size_t)(r0 + l) * 64 + c];
  }
  __syncthreads();
  const int wave = tid >> 5, lane = tid & 31, n = lane & 15, half = lane >> 4;
  // stage 1: x = relu(in @ w_mlp1 + b)   (16x96)@(96x64)
  if (wave < 4) {
    int n0 = wave * 16;
    v8f acc; float bv = b1[n0 + n];
#pragma unroll
    for (int r = 0; r < 8; ++r) acc[r] = bv;
#pragma unroll
    for (int kc = 0; kc < 3; ++kc)
      acc = wmma16(frag_a_lds(&sIn[0][kc * 32], 96), frag_b_t(wt1, 96, n0, kc * 32), acc);
#pragma unroll
    for (int r = 0; r < 8; ++r) {
      float v = acc[r];
      sX[half * 8 + r][n0 + n] = (_Float16)(v > 0.f ? v : 0.f);
    }
  }
  __syncthreads();
  // stage 2: gx = x@Wih+bih, gh = h@Whh+bhh  (24 tiles / 8 waves)
  for (int t = 0; t < 3; ++t) {
    int job = wave * 3 + t;
    const _Float16 (*A)[64] = (job < 12) ? sX : sH;
    const _Float16* Wt = (job < 12) ? wtih : wthh;
    const float* bb = (job < 12) ? bih : bhh;
    int n0 = (job % 12) * 16;
    int gcol = ((job < 12) ? 0 : 192) + n0;
    v8f acc; float bv = bb[n0 + n];
#pragma unroll
    for (int r = 0; r < 8; ++r) acc[r] = bv;
#pragma unroll
    for (int kc = 0; kc < 2; ++kc)
      acc = wmma16(frag_a_lds(&A[0][kc * 32], 64), frag_b_t(Wt, 64, n0, kc * 32), acc);
#pragma unroll
    for (int r = 0; r < 8; ++r) sG[half * 8 + r][gcol + n] = acc[r];
  }
  __syncthreads();
  // gates (PyTorch order r,z,n)
  for (int e = tid; e < 16 * 64; e += 256) {
    int l = e >> 6, c = e & 63;
    float xr = sG[l][c],       hr = sG[l][192 + c];
    float xz = sG[l][64 + c],  hz = sG[l][256 + c];
    float xn = sG[l][128 + c], hn = sG[l][320 + c];
    float rg = 1.f / (1.f + __expf(-(xr + hr)));
    float zg = 1.f / (1.f + __expf(-(xz + hz)));
    float ng = tanhf(xn + rg * hn);
    float hp = h0[(size_t)(r0 + l) * 64 + c];
    hout[(size_t)(r0 + l) * 64 + c] = (1.f - zg) * ng + zg * hp;
  }
}

// ---------------- K2: h @ w_ae1 + b  + batch stats ----------------
__global__ __launch_bounds__(256)
void k_ae1(const float* __restrict__ h, const _Float16* __restrict__ wt,
           const float* __restrict__ bias, float* __restrict__ y1,
           float* __restrict__ sum, float* __restrict__ sqs) {
  __shared__ alignas(16) _Float16 sA[16][64];
  __shared__ alignas(16) float sY[16][128];
  const int r0 = blockIdx.x * 16, tid = threadIdx.x;
  for (int e = tid; e < 1024; e += 256) {
    int l = e >> 6, c = e & 63;
    sA[l][c] = (_Float16)h[(size_t)(r0 + l) * 64 + c];
  }
  __syncthreads();
  const int wave = tid >> 5, lane = tid & 31, n = lane & 15, half = lane >> 4;
  int n0 = wave * 16;
  v8f acc; float bv = bias[n0 + n];
#pragma unroll
  for (int r = 0; r < 8; ++r) acc[r] = bv;
#pragma unroll
  for (int kc = 0; kc < 2; ++kc)
    acc = wmma16(frag_a_lds(&sA[0][kc * 32], 64), frag_b_t(wt, 64, n0, kc * 32), acc);
#pragma unroll
  for (int r = 0; r < 8; ++r) {
    float v = acc[r];
    sY[half * 8 + r][n0 + n] = v;
    y1[(size_t)(r0 + half * 8 + r) * 128 + n0 + n] = v;
  }
  __syncthreads();
  if (tid < 128) {
    float s = 0.f, q = 0.f;
    for (int l = 0; l < 16; ++l) { float v = sY[l][tid]; s += v; q += v * v; }
    atomicAdd(&sum[tid], s);
    atomicAdd(&sqs[tid], q);
  }
}

// ---------------- finalize BN stats -> scale/shift ----------------
__global__ void k_bnfin(const float* __restrict__ sum, const float* __restrict__ sqs,
                        const float* __restrict__ g, const float* __restrict__ be,
                        float inv_cnt, int C, float* __restrict__ sc, float* __restrict__ sh) {
  int c = blockIdx.x * 128 + threadIdx.x;
  if (c < C) {
    float m = sum[c] * inv_cnt;
    float v = sqs[c] * inv_cnt - m * m;
    float is = rsqrtf(v + 1e-5f);
    float s = g[c] * is;
    sc[c] = s;
    sh[c] = be[c] - m * s;
  }
}

// ---------------- K4: lrelu(bn(y1)) @ w_ae2 -> mean / var (weights async-staged in LDS) ------
__global__ __launch_bounds__(256)
void k_ae2(const float* __restrict__ y1, const float* __restrict__ sc, const float* __restrict__ sh,
           const _Float16* __restrict__ wt, const float* __restrict__ bias,
           float* __restrict__ meanB, float* __restrict__ varB) {
  __shared__ alignas(16) _Float16 sA[16][128];
  __shared__ alignas(16) _Float16 sW[128][128];   // this column group's Wt block
  const int r0 = blockIdx.x * 16, cg = blockIdx.y, tid = threadIdx.x;
  stage_w_async(wt + (size_t)cg * 128 * 128, &sW[0][0], 2048);   // 32KB, overlaps A staging
  for (int e = tid; e < 2048; e += 256) {
    int l = e >> 7, c = e & 127;
    float v = y1[(size_t)(r0 + l) * 128 + c] * sc[c] + sh[c];
    sA[l][c] = (_Float16)(v >= 0.f ? v : 0.01f * v);
  }
  wait_async();
  __syncthreads();
  const int wave = tid >> 5, lane = tid & 31, n = lane & 15, half = lane >> 4;
  int n0 = wave * 16;             // local tile inside the column group
  int n0g = cg * 128 + n0;        // global output column base
  v8f acc; float bv = bias[n0g + n];
#pragma unroll
  for (int r = 0; r < 8; ++r) acc[r] = bv;
#pragma unroll
  for (int kc = 0; kc < 4; ++kc)
    acc = wmma16(frag_a_lds(&sA[0][kc * 32], 128), frag_b_t(&sW[0][0], 128, n0, kc * 32), acc);
#pragma unroll
  for (int r = 0; r < 8; ++r) {
    int row = r0 + half * 8 + r, col = n0g + n;
    float v = acc[r];
    if (col < 512) meanB[(size_t)row * 512 + col] = v;
    else           varB[(size_t)row * 512 + col - 512] = fmaxf(__expf(v), 0.002f);
  }
}

// ---------------- K5: awareness = mean + sqrt(var)*noise ----------------
__global__ void k_aware(const float* __restrict__ meanB, const float* __restrict__ varB,
                        const float* __restrict__ noise, float* __restrict__ awB, int total) {
  int i = blockIdx.x * 256 + threadIdx.x;
  if (i < total) awB[i] = meanB[i] + sqrtf(varB[i]) * noise[i];
}

// ---------------- infer-net A staging: row r=(i,b,j) -> [h[b,i] | h[b,j]] ----------------
DEVI void stage_inf_A(const float* __restrict__ h, int r0, _Float16 (*sA)[128]) {
  int tid = threadIdx.x;
  int l = tid >> 4, seg = tid & 15;
  int r = r0 + l;
  int i = r >> 15;          // r / 32768
  int rem = r & 32767;
  int b = rem >> 3, j = rem & 7;
  int cb = seg * 8;
  const float* src = (cb < 64) ? (h + (size_t)(b * 8 + i) * 64 + cb)
                               : (h + (size_t)(b * 8 + j) * 64 + (cb - 64));
#pragma unroll
  for (int u = 0; u < 8; ++u) sA[l][cb + u] = (_Float16)src[u];
}

// ---------------- K6: infer net pre-BN + stats (no store, recomputed later) ----------------
__global__ __launch_bounds__(256)
void k_inf1(const float* __restrict__ h, const _Float16* __restrict__ wt,
            const float* __restrict__ bias, float* __restrict__ sum, float* __restrict__ sqs) {
  __shared__ alignas(16) _Float16 sA[16][128];
  __shared__ alignas(16) float sY[16][128];
  __shared__ alignas(16) _Float16 sW[128][128];
  const int r0 = blockIdx.x * 16, tid = threadIdx.x;
  stage_w_async(wt, &sW[0][0], 2048);
  stage_inf_A(h, r0, sA);
  wait_async();
  __syncthreads();
  const int wave = tid >> 5, lane = tid & 31, n = lane & 15, half = lane >> 4;
  int n0 = wave * 16;
  v8f acc; float bv = bias[n0 + n];
#pragma unroll
  for (int r = 0; r < 8; ++r) acc[r] = bv;
#pragma unroll
  for (int kc = 0; kc < 4; ++kc)
    acc = wmma16(frag_a_lds(&sA[0][kc * 32], 128), frag_b_t(&sW[0][0], 128, n0, kc * 32), acc);
#pragma unroll
  for (int r = 0; r < 8; ++r) sY[half * 8 + r][n0 + n] = acc[r];
  __syncthreads();
  if (tid < 128) {
    float s = 0.f, q = 0.f;
    for (int l = 0; l < 16; ++l) { float v = sY[l][tid]; s += v; q += v * v; }
    atomicAdd(&sum[tid], s);
    atomicAdd(&sqs[tid], q);
  }
}

// ---------------- K8: infer net full + KLD partials ----------------
__global__ __launch_bounds__(256)
void k_inf2(const float* __restrict__ h, const _Float16* __restrict__ wt1,
            const float* __restrict__ b1, const float* __restrict__ sc,
            const float* __restrict__ sh, const _Float16* __restrict__ wt2,
            const float* __restrict__ b2, const float* __restrict__ meanB,
            const float* __restrict__ varB, float* __restrict__ kldpart) {
  __shared__ alignas(16) _Float16 sA[16][128];
  __shared__ alignas(16) float    sY[16][128];
  __shared__ alignas(16) _Float16 sZ[16][128];
  __shared__ alignas(16) float    sP[16][16];
  __shared__ alignas(16) _Float16 sW[128][128];   // staged with wt1, then re-staged with wt2
  const int r0 = blockIdx.x * 16, tid = threadIdx.x;
  stage_w_async(wt1, &sW[0][0], 2048);
  stage_inf_A(h, r0, sA);
  wait_async();
  __syncthreads();
  const int wave = tid >> 5, lane = tid & 31, n = lane & 15, half = lane >> 4;
  const int n0 = wave * 16;
  {
    v8f acc; float bv = b1[n0 + n];
#pragma unroll
    for (int r = 0; r < 8; ++r) acc[r] = bv;
#pragma unroll
    for (int kc = 0; kc < 4; ++kc)
      acc = wmma16(frag_a_lds(&sA[0][kc * 32], 128), frag_b_t(&sW[0][0], 128, n0, kc * 32), acc);
#pragma unroll
    for (int r = 0; r < 8; ++r) sY[half * 8 + r][n0 + n] = acc[r];
  }
  __syncthreads();
  stage_w_async(wt2, &sW[0][0], 2048);   // overlap re-stage with BN+lrelu pass
  for (int e = tid; e < 2048; e += 256) {
    int l = e >> 7, c = e & 127;
    float v = sY[l][c] * sc[c] + sh[c];
    sZ[l][c] = (_Float16)(v >= 0.f ? v : 0.01f * v);
  }
  wait_async();
  __syncthreads();
  {
    v8f acc; float bv = b2[n0 + n];
#pragma unroll
    for (int r = 0; r < 8; ++r) acc[r] = bv;
#pragma unroll
    for (int kc = 0; kc < 4; ++kc)
      acc = wmma16(frag_a_lds(&sZ[0][kc * 32], 128), frag_b_t(&sW[0][0], 128, n0, kc * 32), acc);
#pragma unroll
    for (int r = 0; r < 8; ++r) sY[half * 8 + r][n0 + n] = acc[r];   // ip
  }
  __syncthreads();
  {
    int l = tid >> 4, grp = tid & 15;
    int r = r0 + l;
    int i = r >> 15, rem = r & 32767;
    int b = rem >> 3, j = rem & 7;
    float part = 0.f;
#pragma unroll
    for (int u = 0; u < 4; ++u) {
      int c = grp * 4 + u;
      float im = sY[l][c];
      float iv = fmaxf(__expf(sY[l][64 + c]), 0.002f);
      size_t mi = (size_t)(b * 8 + i) * 512 + j * 64 + c;
      float m = meanB[mi], v = varB[mi];
      float d = m - im;
      part += 0.5f * __logf(iv / v) + 0.5f * (v + d * d) / iv - 0.5f;
    }
    sP[l][grp] = part;
  }
  __syncthreads();
  if (tid < 16) {
    float s = 0.f;
    for (int g = 0; g < 16; ++g) s += sP[tid][g];
    kldpart[r0 + tid] = s;
  }
}

// ---------------- K9: kld reduce over (i, j) ----------------
__global__ void k_kld(const float* __restrict__ kldpart, float* __restrict__ out) {
  int b = blockIdx.x * 256 + threadIdx.x;
  if (b < 4096) {
    float s = 0.f;
    for (int i = 0; i < 8; ++i)
      for (int j = 0; j < 8; ++j)
        s += kldpart[(size_t)i * 32768 + b * 8 + j];
    out[b] = s * (1.f / 4096.f);
  }
}

// ---------------- K10: fused multi-head attention + residual + LN (in-place on awB) ----------
__global__ __launch_bounds__(256)
void k_attn(float* __restrict__ awB,
            const _Float16* __restrict__ wtq, const _Float16* __restrict__ wtk,
            const _Float16* __restrict__ wtv, const _Float16* __restrict__ wtfc,
            const float* __restrict__ gln, const float* __restrict__ bln) {
  __shared__ alignas(16) _Float16 sA[16][64];
  __shared__ alignas(16) _Float16 sQ[16][128], sK[16][128], sV[16][128];
  __shared__ alignas(16) float    sS[4][16][16];
  __shared__ alignas(16) _Float16 sSf[4][16][32];
  __shared__ alignas(16) _Float16 sO[16][128];
  __shared__ alignas(16) float    sR[16][64];
  const int r0 = blockIdx.x * 2;   // two batch rows -> 16 agent-rows
  const int tid = threadIdx.x;
  for (int e = tid; e < 16 * 64; e += 256) {
    int l = e >> 6, c = e & 63;
    int rr = r0 + (l >> 3), ag = l & 7;
    sA[l][c] = (_Float16)awB[(size_t)rr * 512 + ag * 64 + c];
  }
  for (int e = tid; e < 4 * 16 * 16; e += 256) {       // zero K-pad of score tiles
    int hd = e >> 8, rm = e & 255;
    sSf[hd][rm >> 4][16 + (rm & 15)] = (_Float16)0.f;
  }
  __syncthreads();
  const int wave = tid >> 5, lane = tid & 31, n = lane & 15, half = lane >> 4;
  // Q/K/V projections: 24 tiles over 8 waves
  for (int t = 0; t < 3; ++t) {
    int job = wave + t * 8;
    int which = job >> 3;
    int n0 = (job & 7) * 16;
    const _Float16* wt = (which == 0) ? wtq : ((which == 1) ? wtk : wtv);
    _Float16 (*dst)[128] = (which == 0) ? sQ : ((which == 1) ? sK : sV);
    v8f acc;
#pragma unroll
    for (int r = 0; r < 8; ++r) acc[r] = 0.f;
#pragma unroll
    for (int kc = 0; kc < 2; ++kc)
      acc = wmma16(frag_a_lds(&sA[0][kc * 32], 64), frag_b_t(wt, 64, n0, kc * 32), acc);
#pragma unroll
    for (int r = 0; r < 8; ++r) dst[half * 8 + r][n0 + n] = (_Float16)acc[r];
  }
  __syncthreads();
  // scores S = Q_h K_h^T / sqrt(32): heads on waves 0..3
  if (wave < 4) {
    int hd = wave;
    v16h a = frag_a_lds(&sQ[0][hd * 32], 128);
    v16h bf;
    const _Float16* p = &sK[n][hd * 32 + half * 16];
#pragma unroll
    for (int i = 0; i < 16; ++i) bf[i] = p[i];
    v8f acc;
#pragma unroll
    for (int r = 0; r < 8; ++r) acc[r] = 0.f;
    acc = wmma16(a, bf, acc);
    const float scl = 0.17677669529663687f;   // 1/sqrt(32)
#pragma unroll
    for (int r = 0; r < 8; ++r) sS[hd][half * 8 + r][n] = acc[r] * scl;
  }
  __syncthreads();
  // block-diagonal softmax (cross-row blocks forced to zero)
  if (tid < 64) {
    int hd = tid >> 4, l = tid & 15;
    int base = (l < 8) ? 0 : 8;
    float mx = -1e30f;
    for (int c = 0; c < 8; ++c) mx = fmaxf(mx, sS[hd][l][base + c]);
    float ex[8], sm = 0.f;
    for (int c = 0; c < 8; ++c) { ex[c] = __expf(sS[hd][l][base + c] - mx); sm += ex[c]; }
    float inv = 1.f / sm;
    for (int c = 0; c < 16; ++c) sSf[hd][l][c] = (_Float16)0.f;
    for (int c = 0; c < 8; ++c) sSf[hd][l][base + c] = (_Float16)(ex[c] * inv);
  }
  __syncthreads();
  // O = attn @ V : 8 jobs (head, half of dv)
  {
    int hd = wave >> 1, n0 = (wave & 1) * 16;
    v16h a = frag_a_lds(&sSf[hd][0][0], 32);
    v16h bf;
#pragma unroll
    for (int i = 0; i < 16; ++i) {
      int k = half * 16 + i;
      bf[i] = (k < 16) ? sV[k][hd * 32 + n0 + n] : (_Float16)0.f;
    }
    v8f acc;
#pragma unroll
    for (int r = 0; r < 8; ++r) acc[r] = 0.f;
    acc = wmma16(a, bf, acc);
#pragma unroll
    for (int r = 0; r < 8; ++r) sO[half * 8 + r][hd * 32 + n0 + n] = (_Float16)acc[r];
  }
  __syncthreads();
  // fc + residual
  if (wave < 4) {
    int n0 = wave * 16;
    v8f acc;
#pragma unroll
    for (int r = 0; r < 8; ++r) acc[r] = 0.f;
#pragma unroll
    for (int kc = 0; kc < 4; ++kc)
      acc = wmma16(frag_a_lds(&sO[0][kc * 32], 128), frag_b_t(wtfc, 128, n0, kc * 32), acc);
#pragma unroll
    for (int r = 0; r < 8; ++r) {
      int l = half * 8 + r;
      int rr = r0 + (l >> 3), ag = l & 7;
      sR[l][n0 + n] = acc[r] + awB[(size_t)rr * 512 + ag * 64 + n0 + n];
    }
  }
  __syncthreads();
  // LayerNorm(eps=1e-6) per agent row, write back in place
  if (tid < 16) {
    int l = tid, rr = r0 + (l >> 3), ag = l & 7;
    float m = 0.f;
    for (int c = 0; c < 64; ++c) m += sR[l][c];
    m *= (1.f / 64.f);
    float v = 0.f;
    for (int c = 0; c < 64; ++c) { float d = sR[l][c] - m; v += d * d; }
    v *= (1.f / 64.f);
    float is = rsqrtf(v + 1e-6f);
    for (int c = 0; c < 64; ++c)
      awB[(size_t)rr * 512 + ag * 64 + c] = (sR[l][c] - m) * is * gln[c] + bln[c];
  }
}

// ---------------- K11: q_vals = [h | atten] @ w_mlp2 + b ----------------
__global__ __launch_bounds__(32)
void k_final(const float* __restrict__ h, const float* __restrict__ aw,
             const _Float16* __restrict__ wt, const float* __restrict__ bias,
             float* __restrict__ out) {
  __shared__ alignas(16) _Float16 sA[16][32];
  const int r0 = blockIdx.x * 16, tid = threadIdx.x;
  const int lane = tid & 31, n = lane & 15, half = lane >> 4;
  v8f acc; float bv = (n < 14) ? bias[n] : 0.f;
#pragma unroll
  for (int r = 0; r < 8; ++r) acc[r] = bv;
  for (int kc = 0; kc < 18; ++kc) {
    int k0 = kc * 32;
    for (int e = tid; e < 512; e += 32) {
      int l = e >> 5, kk = e & 31;
      int k = k0 + kk;
      float v = (k < 64) ? h[(size_t)(r0 + l) * 64 + k]
                         : aw[(size_t)(r0 + l) * 512 + (k - 64)];
      sA[l][kk] = (_Float16)v;
    }
    __syncthreads();
    acc = wmma16(frag_a_lds(&sA[0][0], 32), frag_b_t(wt, 576, 0, k0), acc);
    __syncthreads();
  }
  if (n < 14) {
#pragma unroll
    for (int r = 0; r < 8; ++r)
      out[(size_t)(r0 + half * 8 + r) * 14 + n] = acc[r];
  }
}

// ---------------- host launch ----------------
extern "C" void kernel_launch(void* const* d_in, const int* in_sizes, int n_in,
                              void* d_out, int out_size, void* d_ws, size_t ws_size,
                              hipStream_t stream) {
  (void)in_sizes; (void)n_in; (void)out_size; (void)ws_size;
  const float* inp   = (const float*)d_in[0];
  const float* h0    = (const float*)d_in[1];
  const float* noise = (const float*)d_in[2];
  const float* w_mlp1 = (const float*)d_in[3];  const float* b_mlp1 = (const float*)d_in[4];
  const float* w_ih  = (const float*)d_in[5];   const float* b_ih   = (const float*)d_in[6];
  const float* w_hh  = (const float*)d_in[7];   const float* b_hh   = (const float*)d_in[8];
  const float* w_ae1 = (const float*)d_in[9];   const float* b_ae1  = (const float*)d_in[10];
  const float* g_ae  = (const float*)d_in[11];  const float* be_ae  = (const float*)d_in[12];
  const float* w_ae2 = (const float*)d_in[13];  const float* b_ae2  = (const float*)d_in[14];
  const float* w_in1 = (const float*)d_in[15];  const float* b_in1  = (const float*)d_in[16];
  const float* g_in  = (const float*)d_in[17];  const float* be_in  = (const float*)d_in[18];
  const float* w_in2 = (const float*)d_in[19];  const float* b_in2  = (const float*)d_in[20];
  const float* w_q   = (const float*)d_in[21];
  const float* w_k   = (const float*)d_in[22];
  const float* w_v   = (const float*)d_in[23];
  const float* w_fc  = (const float*)d_in[24];
  const float* g_ln  = (const float*)d_in[25];  const float* b_ln   = (const float*)d_in[26];
  const float* w_mlp2 = (const float*)d_in[27]; const float* b_mlp2 = (const float*)d_in[28];

  float* out_q   = (float*)d_out;                       // 32768 x 14
  float* out_h   = out_q + (size_t)32768 * 14;          // 32768 x 64
  float* out_kld = out_h + (size_t)32768 * 64;          // 4096

  // workspace bump allocator (256B aligned)
  char* base = (char*)d_ws;
  size_t off = 0;
  auto alloc = [&](size_t nbytes) -> char* {
    char* r = base + off;
    off = (off + nbytes + 255) & ~(size_t)255;
    return r;
  };
  _Float16* wt1  = (_Float16*)alloc((size_t)64 * 96 * 2);
  _Float16* wtih = (_Float16*)alloc((size_t)192 * 64 * 2);
  _Float16* wthh = (_Float16*)alloc((size_t)192 * 64 * 2);
  _Float16* wta1 = (_Float16*)alloc((size_t)128 * 64 * 2);
  _Float16* wta2 = (_Float16*)alloc((size_t)1024 * 128 * 2);
  _Float16* wti1 = (_Float16*)alloc((size_t)128 * 128 * 2);
  _Float16* wti2 = (_Float16*)alloc((size_t)128 * 128 * 2);
  _Float16* wtq  = (_Float16*)alloc((size_t)128 * 64 * 2);
  _Float16* wtk  = (_Float16*)alloc((size_t)128 * 64 * 2);
  _Float16* wtv  = (_Float16*)alloc((size_t)128 * 64 * 2);
  _Float16* wtfc = (_Float16*)alloc((size_t)64 * 128 * 2);
  _Float16* wtm2 = (_Float16*)alloc((size_t)16 * 576 * 2);
  float* stats   = (float*)alloc(1024 * 4);
  // stats: [0]aeSum [128]aeSqs [256]aeSc [384]aeSh [512]inSum [640]inSqs [768]inSc [896]inSh
  float* y1      = (float*)alloc((size_t)32768 * 128 * 4);
  float* meanB   = (float*)alloc((size_t)32768 * 512 * 4);
  float* varB    = (float*)alloc((size_t)32768 * 512 * 4);
  float* awB     = (float*)alloc((size_t)32768 * 512 * 4);
  float* kldpart = (float*)alloc((size_t)262144 * 4);

  hipMemsetAsync(stats, 0, 1024 * 4, stream);

  auto wtprep = [&](const float* W, _Float16* Wt, int K, int N, int Npad) {
    int total = Npad * K;
    k_wt<<<(total + 255) / 256, 256, 0, stream>>>(W, Wt, K, N, Npad);
  };
  wtprep(w_mlp1, wt1, 96, 64, 64);
  wtprep(w_ih, wtih, 64, 192, 192);
  wtprep(w_hh, wthh, 64, 192, 192);
  wtprep(w_ae1, wta1, 64, 128, 128);
  wtprep(w_ae2, wta2, 128, 1024, 1024);
  wtprep(w_in1, wti1, 128, 128, 128);
  wtprep(w_in2, wti2, 128, 128, 128);
  wtprep(w_q, wtq, 64, 128, 128);
  wtprep(w_k, wtk, 64, 128, 128);
  wtprep(w_v, wtv, 64, 128, 128);
  wtprep(w_fc, wtfc, 128, 64, 64);
  wtprep(w_mlp2, wtm2, 576, 14, 16);

  k_gru<<<2048, 256, 0, stream>>>(inp, h0, wt1, b_mlp1, wtih, b_ih, wthh, b_hh, out_h);
  k_ae1<<<2048, 256, 0, stream>>>(out_h, wta1, b_ae1, y1, stats + 0, stats + 128);
  k_bnfin<<<1, 128, 0, stream>>>(stats + 0, stats + 128, g_ae, be_ae,
                                 1.f / 32768.f, 128, stats + 256, stats + 384);
  k_ae2<<<dim3(2048, 8), 256, 0, stream>>>(y1, stats + 256, stats + 384, wta2, b_ae2,
                                           meanB, varB);
  k_aware<<<65536, 256, 0, stream>>>(meanB, varB, noise, awB, 32768 * 512);
  k_inf1<<<16384, 256, 0, stream>>>(out_h, wti1, b_in1, stats + 512, stats + 640);
  k_bnfin<<<1, 128, 0, stream>>>(stats + 512, stats + 640, g_in, be_in,
                                 1.f / 262144.f, 128, stats + 768, stats + 896);
  k_inf2<<<16384, 256, 0, stream>>>(out_h, wti1, b_in1, stats + 768, stats + 896,
                                    wti2, b_in2, meanB, varB, kldpart);
  k_kld<<<16, 256, 0, stream>>>(kldpart, out_kld);
  k_attn<<<16384, 256, 0, stream>>>(awB, wtq, wtk, wtv, wtfc, g_ln, b_ln);
  k_final<<<2048, 32, 0, stream>>>(out_h, awB, wtm2, b_mlp2, out_q);
}
